// hRSSM_86388972192141
// MI455X (gfx1250) — matrix operand order repaired and checked by reading, not compiled, not run on hardware
//
#include <hip/hip_runtime.h>

typedef __attribute__((ext_vector_type(16))) __bf16       v16bf;
typedef __attribute__((ext_vector_type(8)))  float        v8f;
typedef __attribute__((ext_vector_type(8)))  unsigned int v8u;

#define TSTEPS 64
#define NB     256       // batch
#define INP    1024
#define SIZE   4096
#define BLK    8
#define BDIM   512       // per-block hidden
#define C3     12288     // 3*SIZE
#define BCOL   1536      // 3*BDIM

__device__ __forceinline__ unsigned short f2bf(float f) {
    unsigned int u = __float_as_uint(f);
    u += 0x7FFFu + ((u >> 16) & 1u);   // round-to-nearest-even
    return (unsigned short)(u >> 16);
}

// ---------------------------------------------------------------------------
// Prologue 1: elementwise f32 -> bf16 (x)
// ---------------------------------------------------------------------------
__global__ __launch_bounds__(256) void cvt_bf16(
    const float* __restrict__ in, unsigned short* __restrict__ out, int n4)
{
    int i = blockIdx.x * blockDim.x + threadIdx.x;           // float4 units
    if (i >= n4) return;
    const float4 v = ((const float4*)in)[i];
    uint2 p;
    p.x = (unsigned)f2bf(v.x) | ((unsigned)f2bf(v.y) << 16);
    p.y = (unsigned)f2bf(v.z) | ((unsigned)f2bf(v.w) << 16);
    ((uint2*)out)[i] = p;
}

// ---------------------------------------------------------------------------
// Prologue 2: h0 -> h (f32 state) + bf16 shadow
// ---------------------------------------------------------------------------
__global__ __launch_bounds__(256) void init_h(
    const float* __restrict__ h0, float* __restrict__ h,
    unsigned short* __restrict__ hbf, int n)
{
    int i = blockIdx.x * blockDim.x + threadIdx.x;
    if (i >= n) return;
    const float v = h0[i];
    h[i]   = v;
    hbf[i] = f2bf(v);
}

// ---------------------------------------------------------------------------
// Prologue 3: tiled transpose + f32->bf16:  out[c][r] = bf16(in[r][c])
// ---------------------------------------------------------------------------
__global__ __launch_bounds__(256) void transpose_cvt(
    const float* __restrict__ in, unsigned short* __restrict__ out,
    int rows, int cols)
{
    __shared__ float tile[32][33];
    const size_t bofs = (size_t)blockIdx.z * rows * cols;
    const int r0 = blockIdx.y * 32, c0 = blockIdx.x * 32;
    const int tx = threadIdx.x, ty = threadIdx.y;
    #pragma unroll
    for (int p = 0; p < 4; ++p)
        tile[ty + 8 * p][tx] = in[bofs + (size_t)(r0 + ty + 8 * p) * cols + c0 + tx];
    __syncthreads();
    #pragma unroll
    for (int p = 0; p < 4; ++p)
        out[bofs + (size_t)(c0 + ty + 8 * p) * rows + r0 + tx] =
            f2bf(tile[tx][ty + 8 * p]);
}

// ---------------------------------------------------------------------------
// One pipelined GEMM phase with compile-time strides (keeps address math in
// immediate offsets / uniform SGPR adds; no per-step pointer selection).
// ---------------------------------------------------------------------------
template<int LDA, int LDB, int NSTEP>
__device__ __forceinline__ void gemm_phase(
    const unsigned short* __restrict__ Ag,      // tile base, row-major, ld=LDA
    const unsigned short* __restrict__ Bg,      // tile base (pre-transposed), ld=LDB
    unsigned short (*As)[128][32],              // [2][128][32] ping-pong
    unsigned short (*Bs)[128][32],
    v8f (&acc)[4][2],
    int sr, int sc, int waveM, int waveN, int l16, int hi, int& buf)
{
    auto load_regs = [&](int s, uint4* ra, uint4* rb) {
        #pragma unroll
        for (int p = 0; p < 2; ++p) {
            ra[p] = *(const uint4*)(Ag + (size_t)(sr + 64 * p) * LDA + sc + s * 32);
            rb[p] = *(const uint4*)(Bg + (size_t)(sr + 64 * p) * LDB + sc + s * 32);
        }
    };
    auto store_tile = [&](int b, const uint4* ra, const uint4* rb) {
        #pragma unroll
        for (int p = 0; p < 2; ++p) {
            *(uint4*)&As[b][sr + 64 * p][sc] = ra[p];
            *(uint4*)&Bs[b][sr + 64 * p][sc] = rb[p];
        }
    };
    auto compute = [&](int b) {
        v16bf bfr[2];
        #pragma unroll
        for (int nt = 0; nt < 2; ++nt) {
            const unsigned int* bcol = (const unsigned int*)Bs[b][waveN * 32 + nt * 16 + l16];
            v8u ub;
            #pragma unroll
            for (int p = 0; p < 8; ++p)          // B: K halves split by lane half
                ub[p] = bcol[p + hi * 8];
            bfr[nt] = __builtin_bit_cast(v16bf, ub);
        }
        #pragma unroll
        for (int mt = 0; mt < 4; ++mt) {         // stream A fragments one at a time
            const unsigned int* arow = (const unsigned int*)As[b][waveM * 64 + mt * 16 + l16];
            v8u ua;
            #pragma unroll
            for (int p = 0; p < 8; ++p) {        // A: K halves split by lane half
                ua[p]     = arow[p + hi * 8];
            }
            const v16bf af = __builtin_bit_cast(v16bf, ua);
            acc[mt][0] = __builtin_amdgcn_wmma_f32_16x16x32_bf16(
                false, af, false, bfr[0], (short)0, acc[mt][0], false, false);
            acc[mt][1] = __builtin_amdgcn_wmma_f32_16x16x32_bf16(
                false, af, false, bfr[1], (short)0, acc[mt][1], false, false);
        }
    };

    {   // prologue: stage step 0 into current buffer
        __syncthreads();
        uint4 ra[2], rb[2];
        load_regs(0, ra, rb);
        store_tile(buf, ra, rb);
    }
    for (int s = 0; s < NSTEP; ++s) {
        __syncthreads();
        uint4 na[2], nb[2];
        if (s + 1 < NSTEP) load_regs(s + 1, na, nb);   // issue global loads early
        compute(buf);                                  // WMMAs hide load latency
        if (s + 1 < NSTEP) store_tile(buf ^ 1, na, nb);
        buf ^= 1;
    }
}

// ---------------------------------------------------------------------------
// Fused per-timestep GEMM, all-bf16 operands (pre-converted / pre-transposed):
//   parts[n,c] = sum_i x_t[n,i]*Wx[i,c] + sum_j h[n,kBlk*512+j]*Wrec[kBlk,j,c']
// Block tile 128x128, 8 waves, wave tile 64x32.
// ---------------------------------------------------------------------------
__global__ __launch_bounds__(256, 1) void hrssm_gemm(
    const unsigned short* __restrict__ xbf,   // [256][1024]  bf16 slice at t
    const unsigned short* __restrict__ WxT,   // [12288][1024] bf16 (transposed)
    const unsigned short* __restrict__ WrT,   // [8][1536][512] bf16 (transposed)
    const unsigned short* __restrict__ hbf,   // [256][4096]  bf16
    float*                __restrict__ parts) // [256][12288] f32
{
    __shared__ unsigned short As[2][128][32];
    __shared__ unsigned short Bs[2][128][32];

    const int tid   = threadIdx.x;
    const int lane  = tid & 31;
    const int wave  = tid >> 5;
    const int waveM = wave & 1;
    const int waveN = wave >> 1;
    const int l16   = lane & 15;
    const int hi    = lane >> 4;

    const int cTile = blockIdx.x;            // 0..95
    const int mBase = blockIdx.y * 128;      // 0 or 128
    const int cBase = cTile * 128;
    const int kBlk  = cTile / 12;            // 1536 = 12*128
    const int cLoc  = (cTile % 12) * 128;

    const int sr = tid >> 2;                 // staging: 4 threads/row, uint4 each
    const int sc = (tid & 3) * 8;

    v8f acc[4][2] = {};
    int buf = 0;

    // phase 1: x_t[128x1024] * WxT rows (K=1024, 32 steps)
    gemm_phase<INP, INP, 32>(
        xbf + (size_t)mBase * INP,
        WxT + (size_t)cBase * INP,
        As, Bs, acc, sr, sc, waveM, waveN, l16, hi, buf);

    // phase 2: h[128x512 block] * WrecT rows (K=512, 16 steps)
    gemm_phase<SIZE, BDIM, 16>(
        hbf + (size_t)mBase * SIZE + kBlk * BDIM,
        WrT + ((size_t)kBlk * BCOL + cLoc) * BDIM,
        As, Bs, acc, sr, sc, waveM, waveN, l16, hi, buf);

    // epilogue: C/D layout (VGPR v -> M = v + 8*hi, N = l16)
    #pragma unroll
    for (int mt = 0; mt < 4; ++mt) {
        const int row = mBase + waveM * 64 + mt * 16 + hi * 8;
        #pragma unroll
        for (int nt = 0; nt < 2; ++nt) {
            const int col = cBase + waveN * 32 + nt * 16 + l16;
            float* p = parts + (size_t)row * C3 + col;
            #pragma unroll
            for (int v = 0; v < 8; ++v)
                p[(size_t)v * C3] = acc[mt][nt][v];
        }
    }
}

// ---------------------------------------------------------------------------
// Per-row LayerNorm(12288, eps=1e-3) + gating + h update (f32 + bf16 shadow).
// ---------------------------------------------------------------------------
__global__ __launch_bounds__(256) void hrssm_ln_gate(
    const float* __restrict__ parts,
    const float* __restrict__ gamma,
    const float* __restrict__ beta,
    float*       __restrict__ h,
    unsigned short* __restrict__ hbf,
    float*       __restrict__ out_t)
{
    __shared__ float row[C3];
    __shared__ float red[256];
    const int n   = blockIdx.x;
    const int tid = threadIdx.x;

    float s = 0.f, ss = 0.f;
    for (int j = tid; j < C3; j += 256) {
        const float v = parts[(size_t)n * C3 + j];
        row[j] = v;
        s  += v;
        ss += v * v;
    }
    red[tid] = s;
    __syncthreads();
    #pragma unroll
    for (int o = 128; o > 0; o >>= 1) {
        if (tid < o) red[tid] += red[tid + o];
        __syncthreads();
    }
    const float mu = red[0] * (1.0f / C3);
    __syncthreads();
    red[tid] = ss;
    __syncthreads();
    #pragma unroll
    for (int o = 128; o > 0; o >>= 1) {
        if (tid < o) red[tid] += red[tid + o];
        __syncthreads();
    }
    const float var = red[0] * (1.0f / C3) - mu * mu;
    const float inv = rsqrtf(var + 1e-3f);

    for (int j = tid; j < SIZE; j += 256) {
        const float rn = (row[j]          - mu) * inv * gamma[j]          + beta[j];
        const float cn = (row[SIZE + j]   - mu) * inv * gamma[SIZE + j]   + beta[SIZE + j];
        const float zn = (row[2*SIZE + j] - mu) * inv * gamma[2*SIZE + j] + beta[2*SIZE + j];
        const float r  = 1.0f / (1.0f + __expf(-rn));
        const float c  = tanhf(r * cn);
        const float z  = 1.0f / (1.0f + __expf(-(zn - 1.0f)));   // UPDATE_BIAS = -1
        const float hp = h[(size_t)n * SIZE + j];
        const float hn = z * c + (1.0f - z) * hp;
        h[(size_t)n * SIZE + j]     = hn;
        hbf[(size_t)n * SIZE + j]   = f2bf(hn);
        out_t[(size_t)n * SIZE + j] = hn;
    }
}

// ---------------------------------------------------------------------------
extern "C" void kernel_launch(void* const* d_in, const int* in_sizes, int n_in,
                              void* d_out, int out_size, void* d_ws, size_t ws_size,
                              hipStream_t stream) {
    const float* x     = (const float*)d_in[0];  // [64][256][1024]
    const float* h0    = (const float*)d_in[1];  // [256][4096]
    const float* Wx    = (const float*)d_in[2];  // [1024][12288]
    const float* Wrec  = (const float*)d_in[3];  // [8][512][1536]
    const float* gamma = (const float*)d_in[4];  // [12288]
    const float* beta  = (const float*)d_in[5];  // [12288]
    float* out = (float*)d_out;                  // [64][256][4096]

    // workspace layout (all regions 256B aligned)
    unsigned char* w = (unsigned char*)d_ws;
    float*          parts = (float*)(w);                          // 12,582,912 B
    float*          h     = (float*)(w + 12582912u);              //  4,194,304 B
    unsigned short* hbf   = (unsigned short*)(w + 16777216u);     //  2,097,152 B
    unsigned short* xbf   = (unsigned short*)(w + 18874368u);     // 33,554,432 B
    unsigned short* WxT   = (unsigned short*)(w + 52428800u);     // 25,165,824 B
    unsigned short* WrT   = (unsigned short*)(w + 77594624u);     // 12,582,912 B
                                                                  // total ~90.2 MB
    // prologue: convert / transpose everything to bf16 once per call
    {
        const int n4 = TSTEPS * NB * INP / 4;
        cvt_bf16<<<(n4 + 255) / 256, 256, 0, stream>>>(x, xbf, n4);
        init_h<<<(NB * SIZE + 255) / 256, 256, 0, stream>>>(h0, h, hbf, NB * SIZE);
        transpose_cvt<<<dim3(C3 / 32, INP / 32, 1),   dim3(32, 8), 0, stream>>>(Wx,   WxT, INP,  C3);
        transpose_cvt<<<dim3(BCOL / 32, BDIM / 32, BLK), dim3(32, 8), 0, stream>>>(Wrec, WrT, BDIM, BCOL);
    }

    const dim3 gGrid(96, 2);   // 96 col tiles x 2 row tiles of 128x128
    for (int t = 0; t < TSTEPS; ++t) {
        hrssm_gemm<<<gGrid, 256, 0, stream>>>(
            xbf + (size_t)t * NB * INP, WxT, WrT, hbf, parts);
        hrssm_ln_gate<<<NB, 256, 0, stream>>>(
            parts, gamma, beta, h, hbf, out + (size_t)t * NB * SIZE);
    }
}
